// CustomGRU_34239479283915
// MI455X (gfx1250) — compile-verified, matrix operand order checked by reading, not run on hardware
//
#include <hip/hip_runtime.h>
#include <hip/hip_bf16.h>
#include <math.h>

// ---------------------------------------------------------------------------
// CustomGRU forward for MI455X (gfx1250, wave32, WMMA f32_16x16x32_f16)
//   SEQ=2048, BATCH=32, INPUT=256, HIDDEN=256, 3H=768
// Phase 1: f32->f16 conversions (grid-parallel)
// Phase 2: x_proj GEMM (65536x768x256) via WMMA, bias folded, stored f16
// Phase 3: persistent single-workgroup recurrence, 16 waves:
//          - half of W_hh (even 16-col tiles) resident in LDS (~198KB)
//          - other half streamed from the 192MB L2
//          - register double-buffered K loop (loads overlap WMMA issue)
// ---------------------------------------------------------------------------

#define SEQ    2048
#define BATCH  32
#define INPUT  256
#define HIDDEN 256
#define G3     768              // 3*HIDDEN
#define MROWS  (SEQ*BATCH)      // 65536

typedef _Float16 v8h  __attribute__((ext_vector_type(8)));
typedef _Float16 v16h __attribute__((ext_vector_type(16)));
typedef float    v8f  __attribute__((ext_vector_type(8)));

// WMMA f16 operand fragment (A 16x32 or B 32x16), ISA VGPR layout:
// lane L holds 8 contiguous halves at (row0+(L&15), k0+(L>>4)*8) and 8 at +16.
static __device__ __forceinline__ v16h load_frag_g(const _Float16* __restrict__ p,
                                                   int row, int ld, int k, int lane) {
    const _Float16* b0 = p + (size_t)(row + (lane & 15)) * ld + k + (lane >> 4) * 8;
    v8h lo = *(const v8h*)(b0);
    v8h hi = *(const v8h*)(b0 + 16);
    return __builtin_shufflevector(lo, hi, 0,1,2,3,4,5,6,7,8,9,10,11,12,13,14,15);
}
static __device__ __forceinline__ v16h load_frag_l(const _Float16* p,
                                                   int row, int ld, int k, int lane) {
    const _Float16* b0 = p + (size_t)(row + (lane & 15)) * ld + k + (lane >> 4) * 8;
    v8h lo = *(const v8h*)(b0);
    v8h hi = *(const v8h*)(b0 + 16);
    return __builtin_shufflevector(lo, hi, 0,1,2,3,4,5,6,7,8,9,10,11,12,13,14,15);
}

static __device__ __forceinline__ float fast_sigmoid(float x) {
    return 1.0f / (1.0f + __expf(-x));
}
static __device__ __forceinline__ float fast_tanh(float x) {
#if __has_builtin(__builtin_amdgcn_tanhf)
    return __builtin_amdgcn_tanhf(x);
#else
    return tanhf(x);
#endif
}

// -------------------------------- Phase 1 ----------------------------------
__global__ __launch_bounds__(256)
void gru_cvt_f32_to_f16(const float* __restrict__ src, _Float16* __restrict__ dst, int n) {
    for (int i = blockIdx.x * blockDim.x + threadIdx.x; i < n; i += gridDim.x * blockDim.x)
        dst[i] = (_Float16)src[i];
}

// -------------------------------- Phase 2 ----------------------------------
// grid = (512, 12), block = 256 (8 waves). Wave -> one 16-row M tile, 4 N tiles.
// Register double-buffered over K so loads overlap WMMA issue.
__global__ __launch_bounds__(256)
void gru_xproj_wmma(const _Float16* __restrict__ in16,   // [MROWS][256]
                    const _Float16* __restrict__ wih16,  // [768][256]
                    const float*    __restrict__ b_ih,
                    const float*    __restrict__ b_hh,
                    _Float16*       __restrict__ xp16) { // [MROWS][768]
    const int lane = threadIdx.x & 31;
    const int wave = threadIdx.x >> 5;
    const int row0 = (blockIdx.x * 8 + wave) * 16;   // M tile
    const int n0   = blockIdx.y * 64;                // 4 N tiles of 16

    v8f acc[4] = {v8f{}, v8f{}, v8f{}, v8f{}};
    v16h a = load_frag_g(in16, row0, INPUT, 0, lane);
    v16h bf[4];
    #pragma unroll
    for (int t = 0; t < 4; ++t) bf[t] = load_frag_g(wih16, n0 + 16 * t, INPUT, 0, lane);

    #pragma unroll
    for (int ki = 0; ki < 8; ++ki) {
        v16h a_n; v16h bf_n[4];
        if (ki < 7) {
            const int k1 = (ki + 1) * 32;
            a_n = load_frag_g(in16, row0, INPUT, k1, lane);
            #pragma unroll
            for (int t = 0; t < 4; ++t)
                bf_n[t] = load_frag_g(wih16, n0 + 16 * t, INPUT, k1, lane);
        }
        #pragma unroll
        for (int t = 0; t < 4; ++t)
            acc[t] = __builtin_amdgcn_wmma_f32_16x16x32_f16(
                false, a, false, bf[t], (short)0, acc[t], false, false);
        if (ki < 7) {
            a = a_n;
            #pragma unroll
            for (int t = 0; t < 4; ++t) bf[t] = bf_n[t];
        }
    }

    const int hs = lane >> 4;
    #pragma unroll
    for (int t = 0; t < 4; ++t) {
        const int col = n0 + 16 * t + (lane & 15);
        const float bias = b_ih[col] + b_hh[col];
        #pragma unroll
        for (int i = 0; i < 8; ++i)
            xp16[(size_t)(row0 + i + 8 * hs) * G3 + col] = (_Float16)(acc[t][i] + bias);
    }
}

// -------------------------------- Phase 3 ----------------------------------
// 1 workgroup, 512 threads = 16 waves. Wave w: m-tile = w&1, n-tiles grp*6+t.
// Even-parity n-tiles (t = 0,2,4 for every wave) served from LDS W cache;
// odd-parity tiles streamed from L2. Compile-time split, no divergence.
#define H16_LD 264              // padded f16 stride for h
#define G_LD   776              // padded f16 stride for G
#define W_LD   264              // padded f16 stride for LDS W cache rows
#define NC_T   24               // cached tiles (even nt), 24*16 = 384 rows

__global__ __launch_bounds__(512)
void gru_recurrence_wmma(const _Float16* __restrict__ whh16, // [768][256]
                         const _Float16* __restrict__ xp16,  // [MROWS][768]
                         float*          __restrict__ out) { // [SEQ][32][256] + [32][256]
    __shared__ _Float16 wl [NC_T * 16 * W_LD];   // ~198KB: even tiles of W_hh
    __shared__ _Float16 G16[BATCH * G_LD];       // ~48.5KB gate pre-activations
    __shared__ _Float16 h16[BATCH * H16_LD];     // ~16.5KB f16 hidden (A operand)
    __shared__ float    hf [BATCH * HIDDEN];     // 32KB f32 hidden (exact z*h)

    const int tid  = threadIdx.x;
    const int lane = tid & 31;
    const int wave = tid >> 5;
    const int m0   = (wave & 1) * 16;
    const int grp  = wave >> 1;                  // 0..7
    const int hs   = lane >> 4;

    // ---- one-time: preload even W tiles into LDS; zero h ----
    for (int i = tid; i < NC_T * 16 * 32; i += 512) {       // v8h chunks
        const int c  = i >> 9;            // /(16*32)
        const int rr = (i >> 5) & 15;
        const int ck = (i & 31) * 8;
        *(v8h*)&wl[(c * 16 + rr) * W_LD + ck] =
            *(const v8h*)&whh16[((size_t)(c * 32 + rr)) * HIDDEN + ck];
    }
    for (int i = tid; i < BATCH * H16_LD; i += 512) h16[i] = (_Float16)0.0f;
    for (int i = tid; i < BATCH * HIDDEN; i += 512) hf[i] = 0.0f;
    __syncthreads();

    for (int s = 0; s < SEQ; ++s) {
        // ---- G = h @ W_hh^T : 6 tiles per wave, double-buffered K loop ----
        v8f acc[6] = {v8f{}, v8f{}, v8f{}, v8f{}, v8f{}, v8f{}};
        v16h a = load_frag_l(h16, m0, H16_LD, 0, lane);
        v16h bf[6];
        #pragma unroll
        for (int t = 0; t < 6; ++t) {
            const int nt = grp * 6 + t;
            bf[t] = (t & 1) ? load_frag_g(whh16, nt * 16, HIDDEN, 0, lane)
                            : load_frag_l(wl, (nt >> 1) * 16, W_LD, 0, lane);
        }
        #pragma unroll
        for (int ki = 0; ki < 8; ++ki) {
            v16h a_n; v16h bf_n[6];
            if (ki < 7) {
                const int k1 = (ki + 1) * 32;
                a_n = load_frag_l(h16, m0, H16_LD, k1, lane);
                #pragma unroll
                for (int t = 0; t < 6; ++t) {
                    const int nt = grp * 6 + t;
                    bf_n[t] = (t & 1) ? load_frag_g(whh16, nt * 16, HIDDEN, k1, lane)
                                      : load_frag_l(wl, (nt >> 1) * 16, W_LD, k1, lane);
                }
            }
            #pragma unroll
            for (int t = 0; t < 6; ++t)
                acc[t] = __builtin_amdgcn_wmma_f32_16x16x32_f16(
                    false, a, false, bf[t], (short)0, acc[t], false, false);
            if (ki < 7) {
                a = a_n;
                #pragma unroll
                for (int t = 0; t < 6; ++t) bf[t] = bf_n[t];
            }
        }
        #pragma unroll
        for (int t = 0; t < 6; ++t) {
            const int n0 = (grp * 6 + t) * 16;
            #pragma unroll
            for (int i = 0; i < 8; ++i)
                G16[(size_t)(m0 + i + 8 * hs) * G_LD + n0 + (lane & 15)] =
                    (_Float16)acc[t][i];
        }
        __syncthreads();

        // ---- elementwise gates: 8192 units / 512 threads = 16 each ----
        const _Float16* xps = xp16 + (size_t)s * BATCH * G3;
        #pragma unroll
        for (int e = 0; e < 16; ++e) {
            const int idx = tid + e * 512;       // b*256 + j
            const int b   = idx >> 8;
            const int j   = idx & 255;
            const float Gr = (float)G16[(size_t)b * G_LD + j];
            const float Gz = (float)G16[(size_t)b * G_LD + j + 256];
            const float Gn = (float)G16[(size_t)b * G_LD + j + 512];
            const float xr = (float)xps[(size_t)b * G3 + j];
            const float xz = (float)xps[(size_t)b * G3 + j + 256];
            const float xn = (float)xps[(size_t)b * G3 + j + 512];
            const float r  = fast_sigmoid(xr + Gr);
            const float z  = fast_sigmoid(xz + Gz);
            const float n  = fast_tanh(xn + (1.0f + r) * Gn); // quirk: G_n twice
            const float hn = (1.0f - z) * n + z * hf[idx];
            hf[idx]                     = hn;
            h16[(size_t)b * H16_LD + j] = (_Float16)hn;
            out[(size_t)s * (BATCH * HIDDEN) + idx] = hn;
            if (s == SEQ - 1)
                out[(size_t)SEQ * (BATCH * HIDDEN) + idx] = hn;  // h_n tail
            if (s + 1 < SEQ)
                __builtin_prefetch(xps + (size_t)BATCH * G3 + (size_t)b * G3 + j, 0, 3);
        }
        __syncthreads();
    }
}

// ---------------------------------------------------------------------------
extern "C" void kernel_launch(void* const* d_in, const int* in_sizes, int n_in,
                              void* d_out, int out_size, void* d_ws, size_t ws_size,
                              hipStream_t stream) {
    const float* input = (const float*)d_in[0];   // [SEQ][BATCH][INPUT]
    const float* W_ih  = (const float*)d_in[1];   // [768][256]
    const float* W_hh  = (const float*)d_in[2];   // [768][256]
    const float* b_ih  = (const float*)d_in[3];   // [768]
    const float* b_hh  = (const float*)d_in[4];   // [768]
    float* out = (float*)d_out;

    // Workspace layout (f16)
    char* ws = (char*)d_ws;
    _Float16* in16  = (_Float16*)(ws);                                          // 32MB
    _Float16* wih16 = (_Float16*)(ws + (size_t)MROWS * INPUT * 2);              // 384KB
    _Float16* whh16 = (_Float16*)(ws + (size_t)MROWS * INPUT * 2 + 393216);     // 384KB
    _Float16* xp16  = (_Float16*)(ws + (size_t)MROWS * INPUT * 2 + 2 * 393216); // 96MB

    gru_cvt_f32_to_f16<<<4096, 256, 0, stream>>>(input, in16, MROWS * INPUT);
    gru_cvt_f32_to_f16<<<768,  256, 0, stream>>>(W_ih, wih16, G3 * INPUT);
    gru_cvt_f32_to_f16<<<768,  256, 0, stream>>>(W_hh, whh16, G3 * HIDDEN);

    gru_xproj_wmma<<<dim3(512, 12), 256, 0, stream>>>(in16, wih16, b_ih, b_hh, xp16);

    gru_recurrence_wmma<<<1, 512, 0, stream>>>(whh16, xp16, out);
}
